// LatticeLstm_44607530336619
// MI455X (gfx1250) — matrix operand order repaired and checked by reading, not compiled, not run on hardware
//
#include <hip/hip_runtime.h>

// ---------------- constants (match reference) ----------------
constexpr int kIN   = 512;
constexpr int kH    = 512;
constexpr int kG3   = 3 * kH;   // 1536
constexpr int kRING = 8;
constexpr int kW    = 4;

typedef __bf16 v16bf __attribute__((ext_vector_type(16)));
typedef float  v8f   __attribute__((ext_vector_type(8)));

union Frag { v16bf v; uint4 q[2]; };   // 32 bytes: two b128 LDS loads

__device__ __forceinline__ float sigm(float x) { return 1.f / (1.f + __expf(-x)); }

// =====================================================================
// Kernel 1: C[M,N] = A[M,K] * B[K,N] + bias[N]   (fp32 in/out, bf16 WMMA)
// 64x64 macro-tile, K-step 64, per 256-thread block; 8 waves, each wave
// owns a 16x32 strip as two v_wmma_f32_16x16x32_bf16 accumulators.
// LDS layouts chosen so each lane's fragment = contiguous 16B runs
// (ds_load_b128 instead of scalar u16 + v_mov_b16 assembly).
// =====================================================================
__global__ __launch_bounds__(256)
void wmma_gemm_bias(const float* __restrict__ A, const float* __restrict__ B,
                    const float* __restrict__ bias, float* __restrict__ C,
                    int M, int N, int K) {
  __shared__ __bf16 la[64 * 64];   // A tile row-major [m][k], row stride 64
  __shared__ __bf16 lbt[64 * 64];  // B tile TRANSPOSED [n][k], row stride 64

  const int tid  = threadIdx.x;
  const int lane = tid & 31;
  const int wv   = tid >> 5;        // wave 0..7
  const int ar   = (wv >> 1) * 16;  // wave row offset in tile
  const int bc0  = (wv & 1) * 32;   // wave col offset in tile
  const int half = lane >> 4;
  const int l15  = lane & 15;
  const int bm   = blockIdx.y * 64;
  const int bn   = blockIdx.x * 64;

  // staging assignment: 4096 elems per tile, 16 per thread
  const int r  = tid >> 2;          // 0..63 (A row / B k-row)
  const int c0 = (tid & 3) * 16;    // 0,16,32,48

  v8f acc0 = {}; v8f acc1 = {};

  for (int k0 = 0; k0 < K; k0 += 64) {
    // --- stage A 64x64: float4 loads, packed cvt, contiguous bf16 stores ---
    {
      const float4* src = (const float4*)(A + (size_t)(bm + r) * K + k0 + c0);
      #pragma unroll
      for (int e4 = 0; e4 < 4; ++e4) {
        const float4 f = src[e4];
        __bf16* dst = la + r * 64 + c0 + e4 * 4;
        dst[0] = (__bf16)f.x; dst[1] = (__bf16)f.y;
        dst[2] = (__bf16)f.z; dst[3] = (__bf16)f.w;
      }
    }
    // --- stage B 64(k)x64(n) transposed into [n][k] ---
    {
      const float4* src = (const float4*)(B + (size_t)(k0 + r) * N + bn + c0);
      #pragma unroll
      for (int e4 = 0; e4 < 4; ++e4) {
        const float4 f = src[e4];
        const int nn = c0 + e4 * 4;
        lbt[(nn + 0) * 64 + r] = (__bf16)f.x;
        lbt[(nn + 1) * 64 + r] = (__bf16)f.y;
        lbt[(nn + 2) * 64 + r] = (__bf16)f.z;
        lbt[(nn + 3) * 64 + r] = (__bf16)f.w;
      }
    }
    __syncthreads();

    // --- two K=32 WMMA chunks per stage; fragments via ds_load_b128 ---
    #pragma unroll
    for (int kk = 0; kk < 64; kk += 32) {
      Frag af, fb0, fb1;
      // A: lane l15 = row; elems = K {kk+half*8..+8} and {kk+16+half*8..+8}
      const __bf16* pa = la + (ar + l15) * 64 + kk + half * 8;
      af.q[0] = *(const uint4*)(pa);
      af.q[1] = *(const uint4*)(pa + 16);
      // B: lane l15 = col n; elems = K {kk+half*8..+8} and {kk+16+half*8..+8}
      const __bf16* pb0 = lbt + (bc0 + l15) * 64 + kk + half * 8;
      fb0.q[0] = *(const uint4*)(pb0);
      fb0.q[1] = *(const uint4*)(pb0 + 16);
      const __bf16* pb1 = pb0 + 16 * 64;   // second 16-col tile
      fb1.q[0] = *(const uint4*)(pb1);
      fb1.q[1] = *(const uint4*)(pb1 + 16);

      acc0 = __builtin_amdgcn_wmma_f32_16x16x32_bf16(false, af.v, false, fb0.v,
                                                     (short)0, acc0, false, false);
      acc1 = __builtin_amdgcn_wmma_f32_16x16x32_bf16(false, af.v, false, fb1.v,
                                                     (short)0, acc1, false, false);
    }
    __syncthreads();
  }

  // --- store: f32 C/D layout (VGPR r -> row r + 8*half; lane&15 -> col) ---
  const int n0 = bn + bc0 + l15;
  const int n1 = n0 + 16;
  const float bv0 = bias[n0];
  const float bv1 = bias[n1];
  #pragma unroll
  for (int rr = 0; rr < 8; ++rr) {
    const int m = bm + ar + rr + half * 8;
    C[(size_t)m * N + n0] = acc0[rr] + bv0;
    C[(size_t)m * N + n1] = acc1[rr] + bv1;
  }
}

// =====================================================================
// Kernel 2: serial lattice scan. One workgroup, 1024 threads (32 waves).
// State (h, c, gates, wg, wc) in LDS; ring buffers in workspace.
// =====================================================================
__global__ __launch_bounds__(1024)
void lattice_scan(const int* __restrict__ kb_length,
                  const float* __restrict__ Wh,     // [512,1536]
                  const float* __restrict__ Ah,     // [512,512]
                  const float* __restrict__ Wwh,    // [512,1536]
                  const float* __restrict__ preMI,  // [T,1536]  x@Wi + b_mi
                  const float* __restrict__ preA,   // [T,512]   x@Ai + b_alpha
                  const float* __restrict__ preW,   // [T,4,1536] wemb@Wwi + b_w
                  float* __restrict__ ring_c,       // [8,4,512]
                  float* __restrict__ ring_awh,     // [8,4,512]
                  float* __restrict__ out_h,        // [T,512]
                  float* __restrict__ out_c,        // [T,512]
                  int T) {
  __shared__ float sh_h[kH], sh_c[kH];
  __shared__ float sh_gates[kG3];
  __shared__ float sh_wg[kW * kG3];
  __shared__ float sh_wc[kW * kH];
  __shared__ int   sh_rlen[kRING * kW];
  __shared__ int   sh_rstep[kRING];

  const int t = threadIdx.x;

  // ---- init state ----
  if (t < kH) { sh_h[t] = 0.f; sh_c[t] = 0.f; }
  if (t < kRING * kW) sh_rlen[t] = 0;
  if (t < kRING) sh_rstep[t] = -2 * kRING;
  for (int i = t; i < kRING * kW * kH; i += 1024) { ring_c[i] = 0.f; ring_awh[i] = 0.f; }
  __threadfence();
  __syncthreads();

  for (int j = 0; j < T; ++j) {
    // pull next step's precomputed activations toward the WGP (global_prefetch_b8)
    if (j + 1 < T) {
      __builtin_prefetch(&preMI[(size_t)(j + 1) * kG3 + t], 0, 1);
      __builtin_prefetch(&preW[(size_t)(j + 1) * kW * kG3 + t], 0, 1);
    }

    // ---- phase 1: gates = preMI[j] + h @ Wh  (1536 cols over 1024 thr) ----
    {
      float g = preMI[(size_t)j * kG3 + t];
      #pragma unroll 8
      for (int k = 0; k < kH; ++k) g = fmaf(sh_h[k], Wh[(size_t)k * kG3 + t], g);
      sh_gates[t] = g;
      if (t < kG3 - 1024) {
        const int n = 1024 + t;
        float g2 = preMI[(size_t)j * kG3 + n];
        #pragma unroll 8
        for (int k = 0; k < kH; ++k) g2 = fmaf(sh_h[k], Wh[(size_t)k * kG3 + n], g2);
        sh_gates[n] = g2;
      }
    }
    __syncthreads();

    // ---- phase 2: char cell update with ring attention (t < 512) ----
    if (t < kH) {
      const float ig  = sigm(sh_gates[t]);           // i_g = sigmoid
      const float og  = tanhf(sh_gates[kH + t]);     // o_g = tanh (sic)
      const float gg  = sigm(sh_gates[2 * kH + t]);  // g_g = sigmoid (sic)
      const float awi = preA[(size_t)j * kH + t];
      float esum = 0.f, ecsum = 0.f;
      int any = 0;
      #pragma unroll
      for (int s = 0; s < kRING; ++s) {
        const int st = sh_rstep[s];
        #pragma unroll
        for (int w = 0; w < kW; ++w) {
          if (st + sh_rlen[s * kW + w] - 1 == j) {   // uniform predicate
            any = 1;
            const int off = (s * kW + w) * kH + t;
            const float alpha = sigm(awi + ring_awh[off]);
            const float e = __expf(alpha);
            esum += e;
            ecsum = fmaf(e, ring_c[off], ecsum);
          }
        }
      }
      const float ei = __expf(ig);
      const float cw = fmaf(ei, gg, ecsum) / (ei + esum);
      const float cp = fmaf(ig, gg, (1.f - ig) * sh_c[t]);
      const float cn = any ? cw : cp;
      const float hn = og * tanhf(cn);
      sh_c[t] = cn;
      sh_h[t] = hn;
      out_h[(size_t)j * kH + t] = hn;
      out_c[(size_t)j * kH + t] = cn;
    }
    __syncthreads();

    // ---- phase 3: wg = preW[j] + h_new @ Wwh  (4*1536 = 6144 cols) ----
    #pragma unroll
    for (int q = 0; q < 6; ++q) {
      const int idx = t + q * 1024;
      const int w = idx / kG3;
      const int n = idx - w * kG3;
      float g = preW[((size_t)j * kW + w) * kG3 + n];
      #pragma unroll 8
      for (int k = 0; k < kH; ++k) g = fmaf(sh_h[k], Wwh[(size_t)k * kG3 + n], g);
      sh_wg[idx] = g;
    }
    __syncthreads();

    const int s = j & (kRING - 1);

    // ---- phase 4: wc = sig(f)*c_new + sig(i)*tanh(g)  (4*512 vals) ----
    #pragma unroll
    for (int q = 0; q < 2; ++q) {
      const int idx = t + q * 1024;
      const int w  = idx >> 9;
      const int tt = idx & (kH - 1);
      const float fw = sigm(sh_wg[w * kG3 + tt]);
      const float iw = sigm(sh_wg[w * kG3 + kH + tt]);
      const float gw = tanhf(sh_wg[w * kG3 + 2 * kH + tt]);
      const float wcv = fmaf(fw, sh_c[tt], iw * gw);
      sh_wc[w * kH + tt] = wcv;
      ring_c[(s * kW + w) * kH + tt] = wcv;
    }
    __syncthreads();

    // ---- phase 5: awh = wc @ Ah -> ring_awh[s]; ring bookkeeping ----
    #pragma unroll
    for (int q = 0; q < 2; ++q) {
      const int idx = t + q * 1024;
      const int w = idx >> 9;
      const int n = idx & (kH - 1);
      float a = 0.f;
      #pragma unroll 8
      for (int k = 0; k < kH; ++k) a = fmaf(sh_wc[w * kH + k], Ah[(size_t)k * kH + n], a);
      ring_awh[(s * kW + w) * kH + n] = a;
    }
    if (t < kW) sh_rlen[s * kW + t] = kb_length[(size_t)j * kW + t];
    if (t == 0) sh_rstep[s] = j;
    __threadfence();   // ring writes visible across both CUs of the WGP
    __syncthreads();
  }
}

// =====================================================================
extern "C" void kernel_launch(void* const* d_in, const int* in_sizes, int n_in,
                              void* d_out, int out_size, void* d_ws, size_t ws_size,
                              hipStream_t stream) {
  const float* input  = (const float*)d_in[0];   // [T,512]
  const float* kb_emb = (const float*)d_in[1];   // [T,4,512]
  const int*   kb_len = (const int*)d_in[2];     // [T,4]
  const float* Wi   = (const float*)d_in[3];
  const float* Wh   = (const float*)d_in[4];
  const float* b_mi = (const float*)d_in[5];
  const float* Ai   = (const float*)d_in[6];
  const float* Ah   = (const float*)d_in[7];
  const float* b_al = (const float*)d_in[8];
  const float* Wwi  = (const float*)d_in[9];
  const float* Wwh  = (const float*)d_in[10];
  const float* b_w  = (const float*)d_in[11];
  const int T = in_sizes[0] / kIN;               // 1024

  // workspace layout (floats)
  float* ws = (float*)d_ws;
  size_t o = 0;
  float* preMI    = ws + o;  o += (size_t)T * kG3;        // [T,1536]
  float* preA     = ws + o;  o += (size_t)T * kH;         // [T,512]
  float* preW     = ws + o;  o += (size_t)T * kW * kG3;   // [T,4,1536]
  float* ring_c   = ws + o;  o += (size_t)kRING * kW * kH;
  float* ring_awh = ws + o;

  dim3 blk(256);
  // preMI = input @ Wi + b_mi
  wmma_gemm_bias<<<dim3(kG3 / 64, T / 64), blk, 0, stream>>>(
      input, Wi, b_mi, preMI, T, kG3, kIN);
  // preA = input @ Ai + b_alpha
  wmma_gemm_bias<<<dim3(kH / 64, T / 64), blk, 0, stream>>>(
      input, Ai, b_al, preA, T, kH, kIN);
  // preW = kb_embedding(view [T*4,512]) @ Wwi + b_w
  wmma_gemm_bias<<<dim3(kG3 / 64, (T * kW) / 64), blk, 0, stream>>>(
      kb_emb, Wwi, b_w, preW, T * kW, kG3, kIN);

  float* out_h = (float*)d_out;
  float* out_c = (float*)d_out + (size_t)T * kH;
  lattice_scan<<<1, 1024, 0, stream>>>(kb_len, Wh, Ah, Wwh, preMI, preA, preW,
                                       ring_c, ring_awh, out_h, out_c, T);
}
